// AGITransformerArchitecture_19026705122060
// MI455X (gfx1250) — compile-verified, matrix operand order checked by reading, not run on hardware
//
#include <hip/hip_runtime.h>
#include <hip/hip_bf16.h>

// ---------------------------------------------------------------------------
// CDNA5 (gfx1250) wave32 WMMA implementation of the AGI transformer block.
//   * One bf16 conversion pre-pass (weights + activations) halves HBM traffic
//     and removes cvt work from all hot loops.
//   * GEMM + flash attention inner loops run v_wmma_f32_16x16x32_bf16.
//   * LDS staging uses GLOBAL_LOAD_ASYNC_TO_LDS_B128 (ASYNCcnt) via inline
//     asm: per-lane 16B global->LDS copies that bypass VGPRs, finished with
//     s_wait_asynccnt 0 before the workgroup barrier.
// ---------------------------------------------------------------------------

typedef __attribute__((ext_vector_type(16))) __bf16 v16bf;
typedef __attribute__((ext_vector_type(8)))  __bf16 v8bf;
typedef __attribute__((ext_vector_type(8)))  float  v8f;

union BFrag {
    v16bf  v;
    v8bf   h[2];
    __bf16 e[16];
};

// Low 32 bits of a flat shared-aperture address are the LDS byte offset
// (ISA 10.2: LDS aperture maps addr[31:0] -> LDS).
__device__ __forceinline__ unsigned int lds_off(const void* p) {
    return (unsigned int)(unsigned long long)p;
}

// Async 16B global -> LDS copy (VGLOBAL GLOBAL_LOAD_ASYNC_TO_LDS_B128).
// VDST = per-lane LDS byte address, VADDR = per-lane 64-bit global address.
__device__ __forceinline__ void async_copy_b128(unsigned int lds, const void* g) {
    asm volatile("global_load_async_to_lds_b128 %0, %1, off"
                 :: "v"(lds), "v"((unsigned long long)g)
                 : "memory");
}
__device__ __forceinline__ void wait_async0() {
    asm volatile("s_wait_asynccnt 0x0" ::: "memory");
}

// ---------------------------------------------------------------------------
// f32 -> bf16 conversion pre-pass (vectorized 4 elems/thread).
// ---------------------------------------------------------------------------
__global__ __launch_bounds__(256)
void f32_to_bf16_kernel(const float* __restrict__ x, __bf16* __restrict__ y, int n)
{
    const int i = (blockIdx.x * 256 + threadIdx.x) * 4;
    if (i + 3 < n) {
        const float4 v = *(const float4*)(x + i);
        y[i + 0] = (__bf16)v.x; y[i + 1] = (__bf16)v.y;
        y[i + 2] = (__bf16)v.z; y[i + 3] = (__bf16)v.w;
    }
}

// ---------------------------------------------------------------------------
// GEMM: C[M,N] = A[M,K] * W[N,K]^T + bias[N].  A, W bf16; accumulate f32;
// C written as f32 (out_f32 != 0) or bf16.  K must be a multiple of 32.
// Block: 256 threads = 8 waves (2 x 4); tile 128x128x32;
// each wave computes 4x2 16x16 WMMA fragments.
// ---------------------------------------------------------------------------
__global__ __launch_bounds__(256)
void gemm_bf16_kernel(const __bf16* __restrict__ A, const __bf16* __restrict__ W,
                      const float* __restrict__ bias, void* __restrict__ Cout,
                      int M, int N, int K, int out_f32)
{
    constexpr int BM = 128, BN = 128, BK = 32, LS = 40; // padded LDS stride
    __shared__ __align__(16) __bf16 As[BM * LS];
    __shared__ __align__(16) __bf16 Bs[BN * LS];

    const int tid  = threadIdx.x;
    const int w    = tid >> 5;
    const int lane = tid & 31;
    const int half = lane >> 4;
    const int l16  = lane & 15;
    const int wm   = w >> 2;   // 0..1
    const int wn   = w & 3;    // 0..3
    const int mtile = blockIdx.y * BM;
    const int ntile = blockIdx.x * BN;

    v8f acc[4][2];
    const v8f vzero = {0.f, 0.f, 0.f, 0.f, 0.f, 0.f, 0.f, 0.f};
#pragma unroll
    for (int i = 0; i < 4; ++i)
#pragma unroll
        for (int j = 0; j < 2; ++j) acc[i][j] = vzero;

    for (int kt = 0; kt < K; kt += BK) {
        __syncthreads();
        // Prefetch next K tile into L2 while this one is consumed.
        if (kt + BK < K) {
            const int pr = tid >> 1;
            const int pc = (tid & 1) * 16;
            if (mtile + pr < M)
                __builtin_prefetch(A + (size_t)(mtile + pr) * K + kt + BK + pc, 0, 0);
            if (ntile + pr < N)
                __builtin_prefetch(W + (size_t)(ntile + pr) * K + kt + BK + pc, 0, 0);
        }
        // Async-stage A/B tiles (16B per unit, 4 units per thread).
        for (int idx = tid; idx < BM * 4; idx += 256) {
            const int r  = idx >> 2;
            const int c8 = (idx & 3) * 8;
            if (mtile + r < M) {
                async_copy_b128(lds_off(&As[r * LS + c8]),
                                A + (size_t)(mtile + r) * K + kt + c8);
            } else {
#pragma unroll
                for (int t = 0; t < 8; ++t) As[r * LS + c8 + t] = (__bf16)0.f;
            }
            if (ntile + r < N) {
                async_copy_b128(lds_off(&Bs[r * LS + c8]),
                                W + (size_t)(ntile + r) * K + kt + c8);
            } else {
#pragma unroll
                for (int t = 0; t < 8; ++t) Bs[r * LS + c8 + t] = (__bf16)0.f;
            }
        }
        wait_async0();
        __syncthreads();

        // Fragments (ISA 16-bit layout): lane l%16 = row,
        // e[0..7] = K half*8.., e[8..15] = K 16+half*8..
        BFrag af[4], bf[2];
#pragma unroll
        for (int i = 0; i < 4; ++i) {
            const __bf16* p = &As[(wm * 64 + i * 16 + l16) * LS + half * 8];
            af[i].h[0] = *(const v8bf*)(p);
            af[i].h[1] = *(const v8bf*)(p + 16);
        }
#pragma unroll
        for (int j = 0; j < 2; ++j) {
            const __bf16* p = &Bs[(wn * 32 + j * 16 + l16) * LS + half * 8];
            bf[j].h[0] = *(const v8bf*)(p);
            bf[j].h[1] = *(const v8bf*)(p + 16);
        }
#pragma unroll
        for (int i = 0; i < 4; ++i)
#pragma unroll
            for (int j = 0; j < 2; ++j)
                acc[i][j] = __builtin_amdgcn_wmma_f32_16x16x32_bf16(
                    false, af[i].v, false, bf[j].v, (short)0, acc[i][j],
                    false, false);
    }

    // Epilogue (C/D layout: VGPR j -> row j (lanes 0-15) / j+8 (lanes 16-31)).
    if (out_f32) {
        float* C = (float*)Cout;
#pragma unroll
        for (int jn = 0; jn < 2; ++jn) {
            const int col = ntile + wn * 32 + jn * 16 + l16;
            if (col >= N) continue;
            const float bv = bias ? bias[col] : 0.f;
#pragma unroll
            for (int i = 0; i < 4; ++i)
#pragma unroll
                for (int j = 0; j < 8; ++j) {
                    const int row = mtile + wm * 64 + i * 16 + j + 8 * half;
                    if (row < M) C[(size_t)row * N + col] = acc[i][jn][j] + bv;
                }
        }
    } else {
        __bf16* C = (__bf16*)Cout;
#pragma unroll
        for (int jn = 0; jn < 2; ++jn) {
            const int col = ntile + wn * 32 + jn * 16 + l16;
            if (col >= N) continue;
            const float bv = bias ? bias[col] : 0.f;
#pragma unroll
            for (int i = 0; i < 4; ++i)
#pragma unroll
                for (int j = 0; j < 8; ++j) {
                    const int row = mtile + wm * 64 + i * 16 + j + 8 * half;
                    if (row < M) C[(size_t)row * N + col] = (__bf16)(acc[i][jn][j] + bv);
                }
        }
    }
}

// ---------------------------------------------------------------------------
// Flash attention (online softmax), templated on head_dim HD (64 or 256).
// q/k/v/out: bf16, rows of row_stride/out_stride elements, head h at columns
// h*HD..h*HD+HD-1.  modp: optional f32 per-(b,s,h) pre-softmax factor.
// Grid (S/128, H, B); block 256; each wave owns 16 query rows.
// ---------------------------------------------------------------------------
template <int HD>
__global__ __launch_bounds__(256)
void attn_kernel(const __bf16* __restrict__ qp, const __bf16* __restrict__ kp,
                 const __bf16* __restrict__ vp, int row_stride,
                 __bf16* __restrict__ outp, int out_stride,
                 const float* __restrict__ modp, float scale, int S, int H)
{
    constexpr int KVS = HD + 16;   // padded bf16 stride (keeps 16B alignment)
    constexpr int PS  = 40;        // 32 + 8 pad
    __shared__ __align__(16) __bf16 Kt[32 * KVS];
    __shared__ __align__(16) __bf16 Vt[32 * KVS];
    __shared__ __align__(16) __bf16 Pt[8 * 16 * PS];

    const int tid  = threadIdx.x;
    const int w    = tid >> 5;
    const int lane = tid & 31;
    const int half = lane >> 4;
    const int l16  = lane & 15;
    const int b    = blockIdx.z;
    const int h    = blockIdx.y;
    const int qbase = blockIdx.x * 128 + w * 16;
    const size_t bS = (size_t)b * S;

    // Q fragments, cached for the whole K loop (direct bf16 16B loads).
    BFrag qf[HD / 32];
#pragma unroll
    for (int dk = 0; dk < HD / 32; ++dk) {
        const __bf16* p = qp + (bS + qbase + l16) * row_stride + h * HD
                          + dk * 32 + half * 8;
        qf[dk].h[0] = *(const v8bf*)(p);
        qf[dk].h[1] = *(const v8bf*)(p + 16);
    }

    // Per-row pre-softmax factor: scale * sigmoid-gate (if present).
    float rf[8];
#pragma unroll
    for (int j = 0; j < 8; ++j) {
        const int r = qbase + j + 8 * half;
        const float m = modp ? modp[(bS + r) * H + h] : 1.f;
        rf[j] = scale * m;
    }

    float mrow[8], lrow[8];
#pragma unroll
    for (int j = 0; j < 8; ++j) { mrow[j] = -INFINITY; lrow[j] = 0.f; }
    v8f acc[HD / 16];
    const v8f vzero = {0.f, 0.f, 0.f, 0.f, 0.f, 0.f, 0.f, 0.f};
#pragma unroll
    for (int dt = 0; dt < HD / 16; ++dt) acc[dt] = vzero;

    for (int kb = 0; kb < S; kb += 32) {
        __syncthreads();
        // Async-stage K/V 32 x HD bf16 tiles into LDS (16B units).
        constexpr int NU = 32 * HD / 8;
        for (int u = tid; u < NU; u += 256) {
            const int r = u / (HD / 8);
            const int c = (u % (HD / 8)) * 8;
            const size_t goff = (bS + kb + r) * row_stride + h * HD + c;
            async_copy_b128(lds_off(&Kt[r * KVS + c]), kp + goff);
            async_copy_b128(lds_off(&Vt[r * KVS + c]), vp + goff);
        }
        wait_async0();
        __syncthreads();

        // Scores: two 16x16 tiles (keys kb..kb+15, kb+16..kb+31).
        v8f s0 = vzero, s1 = vzero;
#pragma unroll
        for (int dk = 0; dk < HD / 32; ++dk) {
            BFrag k0, k1;
            const __bf16* p0 = &Kt[l16 * KVS + dk * 32 + half * 8];
            const __bf16* p1 = &Kt[(16 + l16) * KVS + dk * 32 + half * 8];
            k0.h[0] = *(const v8bf*)(p0);  k0.h[1] = *(const v8bf*)(p0 + 16);
            k1.h[0] = *(const v8bf*)(p1);  k1.h[1] = *(const v8bf*)(p1 + 16);
            s0 = __builtin_amdgcn_wmma_f32_16x16x32_bf16(
                false, qf[dk].v, false, k0.v, (short)0, s0, false, false);
            s1 = __builtin_amdgcn_wmma_f32_16x16x32_bf16(
                false, qf[dk].v, false, k1.v, (short)0, s1, false, false);
        }

        // Online softmax; P transposed through a per-wave LDS patch.
        float alpha[8];
#pragma unroll
        for (int j = 0; j < 8; ++j) {
            const float sm0 = s0[j] * rf[j];
            const float sm1 = s1[j] * rf[j];
            float mx = fmaxf(sm0, sm1);
            mx = fmaxf(mx, __shfl_xor(mx, 1, 32));
            mx = fmaxf(mx, __shfl_xor(mx, 2, 32));
            mx = fmaxf(mx, __shfl_xor(mx, 4, 32));
            mx = fmaxf(mx, __shfl_xor(mx, 8, 32));
            const float mnew = fmaxf(mrow[j], mx);
            alpha[j] = __expf(mrow[j] - mnew);
            mrow[j] = mnew;
            const float p0 = __expf(sm0 - mnew);
            const float p1 = __expf(sm1 - mnew);
            __bf16* pr = &Pt[(w * 16 + j + 8 * half) * PS];
            pr[l16]      = (__bf16)p0;
            pr[16 + l16] = (__bf16)p1;
            float ps = p0 + p1;
            ps += __shfl_xor(ps, 1, 32);
            ps += __shfl_xor(ps, 2, 32);
            ps += __shfl_xor(ps, 4, 32);
            ps += __shfl_xor(ps, 8, 32);
            lrow[j] = lrow[j] * alpha[j] + ps;
        }
#pragma unroll
        for (int dt = 0; dt < HD / 16; ++dt)
#pragma unroll
            for (int j = 0; j < 8; ++j) acc[dt][j] *= alpha[j];

        // P back as an A-fragment (same wave wrote Pt; DS is in-order).
        BFrag pf;
        const __bf16* pb = &Pt[(w * 16 + l16) * PS + half * 8];
        pf.h[0] = *(const v8bf*)(pb);
        pf.h[1] = *(const v8bf*)(pb + 16);

        // PV accumulate: B fragment column = output dim, K = key index.
#pragma unroll
        for (int dt = 0; dt < HD / 16; ++dt) {
            BFrag vf;
#pragma unroll
            for (int i = 0; i < 8; ++i) {
                vf.e[i]     = Vt[(half * 8 + i) * KVS + dt * 16 + l16];
                vf.e[i + 8] = Vt[(16 + half * 8 + i) * KVS + dt * 16 + l16];
            }
            acc[dt] = __builtin_amdgcn_wmma_f32_16x16x32_bf16(
                false, pf.v, false, vf.v, (short)0, acc[dt], false, false);
        }
    }

    // Normalize and store (bf16).
#pragma unroll
    for (int dt = 0; dt < HD / 16; ++dt)
#pragma unroll
        for (int j = 0; j < 8; ++j) {
            const int r = qbase + j + 8 * half;
            outp[(bS + r) * out_stride + h * HD + dt * 16 + l16] =
                (__bf16)(acc[dt][j] / lrow[j]);
        }
}

// ---------------------------------------------------------------------------
// Small kernels
// ---------------------------------------------------------------------------
__global__ void aw_kernel(const float* __restrict__ cv,
                          const float* __restrict__ aww,
                          const float* __restrict__ awb,
                          float* __restrict__ aw)
{
    const int n = threadIdx.x;
    if (n < 16) {
        float s = awb[n];
#pragma unroll
        for (int k2 = 0; k2 < 16; ++k2) s += cv[k2] * aww[n * 16 + k2];
        aw[n] = s;
    }
}

__global__ void mod_kernel(const float* __restrict__ gate,
                           const float* __restrict__ aw,
                           float* __restrict__ mod, int n)
{
    const int i = blockIdx.x * 256 + threadIdx.x;
    if (i < n) {
        const float x = gate[i] + aw[i & 15];
        mod[i] = 1.f / (1.f + __expf(-x));
    }
}

__global__ void blend_bf16_kernel(const __bf16* x, const __bf16* y, __bf16* o,
                                  float a0, float a1, int n)
{
    const int i = blockIdx.x * 256 + threadIdx.x;
    if (i < n) o[i] = (__bf16)(a0 * (float)x[i] + a1 * (float)y[i]);
}

// ---------------------------------------------------------------------------
// Orchestration
// ---------------------------------------------------------------------------
extern "C" void kernel_launch(void* const* d_in, const int* in_sizes, int n_in,
                              void* d_out, int out_size, void* d_ws, size_t ws_size,
                              hipStream_t stream)
{
    constexpr int Bb = 2, S = 2048, D = 1024, H = 16;
    constexpr int M = Bb * S;                    // 4096 rows

    const float* hs  = (const float*)d_in[0];
    const float* cv  = (const float*)d_in[1];
    const float* wq  = (const float*)d_in[2];
    const float* bq  = (const float*)d_in[3];
    const float* wk  = (const float*)d_in[4];
    const float* bk  = (const float*)d_in[5];
    const float* wv  = (const float*)d_in[6];
    const float* bv  = (const float*)d_in[7];
    const float* gw  = (const float*)d_in[8];
    const float* gb  = (const float*)d_in[9];
    const float* aww = (const float*)d_in[10];
    const float* awb = (const float*)d_in[11];
    const float* ciw = (const float*)d_in[12];
    const float* cib = (const float*)d_in[13];
    const float* cow = (const float*)d_in[14];
    const float* cob = (const float*)d_in[15];
    const float* miw = (const float*)d_in[16];
    const float* mib = (const float*)d_in[17];
    const float* mow = (const float*)d_in[18];
    const float* mob = (const float*)d_in[19];
    const float* ow  = (const float*)d_in[20];
    const float* ob  = (const float*)d_in[21];

    // bf16 region first, f32 region after (all counts even -> aligned).
    __bf16* bws = (__bf16*)d_ws;
    size_t e = 0;
    __bf16* hsb  = bws + e; e += (size_t)M * D;
    __bf16* wqb  = bws + e; e += (size_t)D * D;
    __bf16* wkb  = bws + e; e += (size_t)D * D;
    __bf16* wvb  = bws + e; e += (size_t)D * D;
    __bf16* gwb  = bws + e; e += (size_t)H * D;
    __bf16* ciwb = bws + e; e += (size_t)3 * D * D;
    __bf16* cowb = bws + e; e += (size_t)D * D;
    __bf16* miwb = bws + e; e += (size_t)3 * D * D;
    __bf16* mowb = bws + e; e += (size_t)D * D;
    __bf16* owb  = bws + e; e += (size_t)D * D;
    __bf16* Ab   = bws + e; e += (size_t)M * 3 * D;   // qkv / cqkv / mqkv
    __bf16* ctxb = bws + e; e += (size_t)M * D;
    __bf16* attb = bws + e; e += (size_t)M * D;
    __bf16* projb= bws + e; e += (size_t)M * D;
    float* fws  = (float*)(bws + e);
    float* gate = fws;
    float* mod  = fws + (size_t)M * H;
    float* aw   = fws + (size_t)2 * M * H;

    __bf16* qb  = Ab;
    __bf16* kb2 = Ab + (size_t)M * D;
    __bf16* vb  = Ab + (size_t)M * D * 2;

    const dim3 blk(256);
    const dim3 gD((D + 127) / 128, (M + 127) / 128);
    const dim3 g3D((3 * D + 127) / 128, (M + 127) / 128);
    const dim3 gGate(1, (M + 127) / 128);
    const int  nElem = M * D;
    const dim3 gEl((nElem + 255) / 256);
    auto cvt_grid = [](int n) { return dim3((n / 4 + 255) / 256); };

    // --- bf16 conversion pre-pass (weights + activations) ---
    f32_to_bf16_kernel<<<cvt_grid(M * D), blk, 0, stream>>>(hs, hsb, M * D);
    f32_to_bf16_kernel<<<cvt_grid(D * D), blk, 0, stream>>>(wq, wqb, D * D);
    f32_to_bf16_kernel<<<cvt_grid(D * D), blk, 0, stream>>>(wk, wkb, D * D);
    f32_to_bf16_kernel<<<cvt_grid(D * D), blk, 0, stream>>>(wv, wvb, D * D);
    f32_to_bf16_kernel<<<cvt_grid(H * D), blk, 0, stream>>>(gw, gwb, H * D);
    f32_to_bf16_kernel<<<cvt_grid(3 * D * D), blk, 0, stream>>>(ciw, ciwb, 3 * D * D);
    f32_to_bf16_kernel<<<cvt_grid(D * D), blk, 0, stream>>>(cow, cowb, D * D);
    f32_to_bf16_kernel<<<cvt_grid(3 * D * D), blk, 0, stream>>>(miw, miwb, 3 * D * D);
    f32_to_bf16_kernel<<<cvt_grid(D * D), blk, 0, stream>>>(mow, mowb, D * D);
    f32_to_bf16_kernel<<<cvt_grid(D * D), blk, 0, stream>>>(ow, owb, D * D);

    // --- main QKV projections ---
    gemm_bf16_kernel<<<gD, blk, 0, stream>>>(hsb, wqb, bq, qb,  M, D, D, 0);
    gemm_bf16_kernel<<<gD, blk, 0, stream>>>(hsb, wkb, bk, kb2, M, D, D, 0);
    gemm_bf16_kernel<<<gD, blk, 0, stream>>>(hsb, wvb, bv, vb,  M, D, D, 0);

    // --- consciousness gate modulation (f32 gate/mod) ---
    gemm_bf16_kernel<<<gGate, blk, 0, stream>>>(hsb, gwb, gb, gate, M, H, D, 1);
    aw_kernel<<<1, 32, 0, stream>>>(cv, aww, awb, aw);
    mod_kernel<<<(M * H + 255) / 256, 256, 0, stream>>>(gate, aw, mod, M * H);

    // --- main attention: 16 heads, hd=64, modulated pre-softmax scores ---
    attn_kernel<64><<<dim3(S / 128, 16, Bb), blk, 0, stream>>>(
        qb, kb2, vb, D, ctxb, D, mod, 0.125f, S, 16);

    // --- causal MHA (4 heads, hd=256) ---
    gemm_bf16_kernel<<<g3D, blk, 0, stream>>>(hsb, ciwb, cib, Ab, M, 3 * D, D, 0);
    attn_kernel<256><<<dim3(S / 128, 4, Bb), blk, 0, stream>>>(
        Ab, Ab + D, Ab + 2 * D, 3 * D, attb, D, nullptr, 0.0625f, S, 4);
    gemm_bf16_kernel<<<gD, blk, 0, stream>>>(attb, cowb, cob, projb, M, D, D, 0);
    blend_bf16_kernel<<<gEl, blk, 0, stream>>>(ctxb, projb, ctxb, 0.1f, 0.9f, nElem);

    // --- metacognitive MHA (4 heads, hd=256) ---
    gemm_bf16_kernel<<<g3D, blk, 0, stream>>>(ctxb, miwb, mib, Ab, M, 3 * D, D, 0);
    attn_kernel<256><<<dim3(S / 128, 4, Bb), blk, 0, stream>>>(
        Ab, Ab + D, Ab + 2 * D, 3 * D, attb, D, nullptr, 0.0625f, S, 4);
    gemm_bf16_kernel<<<gD, blk, 0, stream>>>(attb, mowb, mob, projb, M, D, D, 0);
    blend_bf16_kernel<<<gEl, blk, 0, stream>>>(ctxb, projb, ctxb, 0.85f, 0.15f, nElem);

    // --- final output projection (f32 out) ---
    gemm_bf16_kernel<<<gD, blk, 0, stream>>>(ctxb, owb, ob, d_out, M, D, D, 1);

    (void)in_sizes; (void)n_in; (void)out_size; (void)ws_size;
}
